// ImplicitQueryAttention_26242250178730
// MI455X (gfx1250) — matrix operand order repaired and checked by reading, not compiled, not run on hardware
//
#include <hip/hip_runtime.h>
#include <hip/hip_bf16.h>
#include <stdint.h>

#define B_ 2
#define N_ 4096
#define D_ 512
#define ITERS_ 20

typedef __attribute__((ext_vector_type(16))) __bf16 v16bf;
typedef __attribute__((ext_vector_type(8)))  float  v8f;
typedef __attribute__((ext_vector_type(4))) unsigned int u32x4;
typedef __attribute__((ext_vector_type(8))) int i32x8;
typedef __attribute__((ext_vector_type(4))) int i32x4;

#if defined(__has_builtin)
#if __has_builtin(__builtin_amdgcn_tensor_load_to_lds)
#define HAVE_TDM 1
#endif
#endif

union ABu { v16bf v; uint4 u[2]; };

__device__ __forceinline__ v8f zerov() { v8f z = {}; return z; }

__device__ __forceinline__ unsigned short f2bf(float f) {
  unsigned int u = __float_as_uint(f);
  u += 0x7FFFu + ((u >> 16) & 1u);   // round-to-nearest-even
  return (unsigned short)(u >> 16);
}

// A fragment (16x32 bf16) from row-major source with leading dim `ld` (halves).
// CDNA5 layout: lane m=lane&15; lanes<16 hold K {0..7,16..23}, lanes>=16 hold K {8..15,24..31}.
__device__ __forceinline__ v16bf frag_a_rowmajor(const unsigned short* p0, size_t ld, int k0, int lane) {
  const int m = lane & 15, h = lane >> 4;
  const unsigned short* p = p0 + (size_t)m * ld + (size_t)(k0 + 8 * h);
  ABu t;
  t.u[0] = *(const uint4*)p;
  t.u[1] = *(const uint4*)(p + 16);
  return t.v;
}

// B fragment (32x16 bf16) for NT GEMM: B[k][n] = Src[n][k] with Src row-major,
// so a lane (column n) loads K {0..7,16..23} (lanes<16) or {8..15,24..31} (lanes>=16) from Src row n.
__device__ __forceinline__ v16bf frag_b_nt(const unsigned short* p0, size_t ld, int k0, int lane) {
  const int n = lane & 15, h = lane >> 4;
  const unsigned short* p = p0 + (size_t)n * ld + (size_t)(k0 + 8 * h);
  ABu t;
  t.u[0] = *(const uint4*)p;
  t.u[1] = *(const uint4*)(p + 16);
  return t.v;
}

__device__ __forceinline__ v8f wmma_bf16(v16bf a, v16bf b, v8f c) {
  return __builtin_amdgcn_wmma_f32_16x16x32_bf16(false, a, false, b, (short)0, c, false, false);
}

// ---------------------------------------------------------------------------
// prep: f32 -> bf16 of x (and transposed copy), W_Q, W_K
// ---------------------------------------------------------------------------
__global__ void __launch_bounds__(256) prep_kernel(
    const float* __restrict__ x, const float* __restrict__ wq, const float* __restrict__ wk,
    unsigned short* __restrict__ Xb, unsigned short* __restrict__ XT,
    unsigned short* __restrict__ WQb, unsigned short* __restrict__ WKb) {
  const size_t total = (size_t)B_ * N_ * D_;
  const size_t dd = (size_t)D_ * D_;
  for (size_t i = (size_t)blockIdx.x * blockDim.x + threadIdx.x; i < total;
       i += (size_t)gridDim.x * blockDim.x) {
    unsigned short bv = f2bf(x[i]);
    Xb[i] = bv;
    size_t b = i / ((size_t)N_ * D_);
    size_t rem = i - b * (size_t)N_ * D_;
    size_t n = rem / D_, d = rem - n * D_;
    XT[(b * D_ + d) * N_ + n] = bv;
    if (i < dd) {
      WQb[i] = f2bf(wq[i]);
      WKb[i] = f2bf(wk[i]);
    }
  }
}

// ---------------------------------------------------------------------------
// proj: Out[n,e] = sum_d Y[n,d] * W[e,d]   (NT GEMM, bf16 in, bf16 out)
// ---------------------------------------------------------------------------
__global__ void __launch_bounds__(256) proj_kernel(
    const unsigned short* __restrict__ Y,   // [B*N][D] bf16
    const unsigned short* __restrict__ W,   // [D][D]   bf16
    unsigned short* __restrict__ Out) {     // [B*N][D] bf16
  const int lane = threadIdx.x & 31;
  const int wave = threadIdx.x >> 5;
  const int tile = blockIdx.x * 8 + wave;
  const int e0 = (tile & 31) * 16;          // D_/16 = 32 column tiles
  const int r0 = (tile >> 5) * 16;          // row tile over B*N
  v8f acc = zerov();
#pragma unroll 4
  for (int ks = 0; ks < D_ / 32; ++ks) {
    v16bf a = frag_a_rowmajor(Y + (size_t)r0 * D_, D_, ks * 32, lane);
    v16bf bb = frag_b_nt(W + (size_t)e0 * D_, D_, ks * 32, lane);
    acc = wmma_bf16(a, bb, acc);
  }
  const int n = lane & 15, hi = lane >> 4;
#pragma unroll
  for (int r = 0; r < 8; ++r) {
    int row = r0 + r + 8 * hi;
    Out[(size_t)row * D_ + e0 + n] = f2bf(acc[r]);
  }
}

// ---------------------------------------------------------------------------
// flash: y[b, i, :] = softmax_m( K[b,i,:] . Q[b,m,:] ) @ x[b, m, :]
// Block = 128 threads (4 waves) per 16-row K-tile; online softmax over m in
// chunks of 128; each wave owns a 128-wide d-slice of the fp32 accumulator.
// K-tile is staged into LDS by the Tensor Data Mover (with HW row padding).
// ---------------------------------------------------------------------------
#define LDA_ 520   // 512 + 8 halves pad (16B): matches TDM pad_interval/amount
#define LDP_ 136   // 128 + 8 halves pad

__global__ void __launch_bounds__(128) flash_kernel(
    const unsigned short* __restrict__ Kb,   // [B][N][D] bf16
    const unsigned short* __restrict__ Qb,   // [B][N][D] bf16
    const unsigned short* __restrict__ XT,   // [B][D][N] bf16 (x transposed)
    unsigned short* __restrict__ Yb,         // [B][N][D] bf16
    float* __restrict__ outF) {              // optional fp32 output (last iter)
  __shared__ unsigned short sA[16 * LDA_];   // K tile 16 x 512 (padded rows)
  __shared__ unsigned short sP[16 * LDP_];   // P tile 16 x 128
  __shared__ float wmaxs[4][16];
  __shared__ float wsums[4][16];

  const int lane = threadIdx.x & 31;
  const int wave = threadIdx.x >> 5;
  const int mt = blockIdx.x & (N_ / 16 - 1);
  const int b = blockIdx.x >> 8;             // N_/16 == 256

#ifdef HAVE_TDM
  // ---- TDM: DMA the K tile (2D tensor, 2-byte elems, 512x16 tile) into LDS,
  // hardware-padding each 256-DWORD row with 4 DWORDs -> 520-half LDS stride.
  if (wave == 0) {
    unsigned long long ga =
        (unsigned long long)(uintptr_t)(Kb + ((size_t)b * N_ + (size_t)mt * 16) * D_);
    unsigned int lds = (unsigned int)(uintptr_t)(void*)&sA[0];  // low 32b = LDS offset
    u32x4 g0;
    g0[0] = 1u;                         // count=1, user descriptor
    g0[1] = lds;                        // lds_addr
    g0[2] = (unsigned int)ga;           // global_addr[31:0]
    g0[3] = (unsigned int)((ga >> 32) & 0x01FFFFFFu) | (2u << 30);  // addr[56:32], type=2
    i32x8 g1;
    g1[0] = (int)((1u << 16) |          // data_size = 2 bytes
                  (1u << 20) |          // pad_enable
                  (7u << 22) |          // pad_interval = 256 DWORDs (one row)
                  (3u << 25));          // pad_amount   = 4 DWORDs (8 halves)
    g1[1] = (int)(512u << 16);          // tensor_dim0 = 512 (low 16 bits)
    g1[2] = (int)(16u << 16);           // tensor_dim1 = 16
    g1[3] = (int)(512u << 16);          // tile_dim0 = 512
    g1[4] = 16;                         // tile_dim1 = 16
    g1[5] = 512;                        // tensor_dim0_stride = 512
    g1[6] = 0;
    g1[7] = 0;
    i32x4 gz = {0, 0, 0, 0};
#if __clang_major__ >= 23
    i32x8 gz8 = {0, 0, 0, 0, 0, 0, 0, 0};
    __builtin_amdgcn_tensor_load_to_lds(g0, g1, gz, gz, gz8, 0);
#else
    __builtin_amdgcn_tensor_load_to_lds(g0, g1, gz, gz, 0);
#endif
    __builtin_amdgcn_s_wait_tensorcnt(0);
  }
  __syncthreads();
#else
  {
    const unsigned short* src = Kb + ((size_t)b * N_ + (size_t)mt * 16) * D_;
    int base = threadIdx.x * 64;
    int row = base >> 9;
    int col = base & 511;
#pragma unroll
    for (int j = 0; j < 4; ++j) {
      uint4 v = *(const uint4*)(src + (size_t)row * D_ + col + j * 8);
      *(uint4*)(&sA[row * LDA_ + col + j * 8]) = v;
    }
  }
  __syncthreads();
#endif

  const int n = lane & 15, hi = lane >> 4;
  v8f O[8];
#pragma unroll
  for (int f = 0; f < 8; ++f) O[f] = zerov();
  float gmax[8], gsum[8];
#pragma unroll
  for (int r = 0; r < 8; ++r) { gmax[r] = -3.0e38f; gsum[r] = 0.f; }

  for (int c = 0; c < N_ / 128; ++c) {
    const unsigned short* q0 = Qb + ((size_t)b * N_ + (size_t)c * 128 + wave * 32) * D_;

    // ---- prefetch next chunk's Q rows and xT rows (global_prefetch_b8)
    if (c + 1 < N_ / 128) {
      const unsigned short* qn = q0 + (size_t)128 * D_;
      __builtin_prefetch(qn + (size_t)lane * D_, 0, 1);          // 32 rows / wave
      const unsigned short* xn =
          XT + ((size_t)b * D_ + wave * 128 + (lane & 7) * 16 + (lane >> 3)) * N_ +
          (size_t)(c + 1) * 128;
      __builtin_prefetch(xn, 0, 1);
    }

    // ---- scores: S (16 rows x 32 q-cols for this wave), f32 accum
    v8f S0 = zerov(), S1 = zerov();
#pragma unroll 4
    for (int ks = 0; ks < D_ / 32; ++ks) {
      v16bf a = frag_a_rowmajor(sA, LDA_, ks * 32, lane);
      v16bf b0 = frag_b_nt(q0, D_, ks * 32, lane);
      S0 = wmma_bf16(a, b0, S0);
      v16bf b1 = frag_b_nt(q0 + (size_t)16 * D_, D_, ks * 32, lane);
      S1 = wmma_bf16(a, b1, S1);
    }
    // ---- per-row max over this wave's 32 columns (cross-lane, width 16)
    float lmax[8];
#pragma unroll
    for (int r = 0; r < 8; ++r) lmax[r] = fmaxf(S0[r], S1[r]);
#pragma unroll
    for (int m = 1; m < 16; m <<= 1)
#pragma unroll
      for (int r = 0; r < 8; ++r)
        lmax[r] = fmaxf(lmax[r], __shfl_xor(lmax[r], m, 32));
    if (n == 0) {
#pragma unroll
      for (int r = 0; r < 8; ++r) wmaxs[wave][r + 8 * hi] = lmax[r];
    }
    __syncthreads();
    float alpha[8];
#pragma unroll
    for (int r = 0; r < 8; ++r) {
      int row = r + 8 * hi;
      float nm = gmax[r];
      nm = fmaxf(nm, wmaxs[0][row]);
      nm = fmaxf(nm, wmaxs[1][row]);
      nm = fmaxf(nm, wmaxs[2][row]);
      nm = fmaxf(nm, wmaxs[3][row]);
      alpha[r] = __expf(gmax[r] - nm);
      gmax[r] = nm;
    }
    // ---- P = exp(S - max): bf16 into LDS, local row sums
    float lsum[8];
#pragma unroll
    for (int r = 0; r < 8; ++r) {
      float p0 = __expf(S0[r] - gmax[r]);
      float p1 = __expf(S1[r] - gmax[r]);
      lsum[r] = p0 + p1;
      int row = r + 8 * hi;
      sP[row * LDP_ + wave * 32 + n] = f2bf(p0);
      sP[row * LDP_ + wave * 32 + 16 + n] = f2bf(p1);
    }
#pragma unroll
    for (int m = 1; m < 16; m <<= 1)
#pragma unroll
      for (int r = 0; r < 8; ++r)
        lsum[r] += __shfl_xor(lsum[r], m, 32);
    if (n == 0) {
#pragma unroll
      for (int r = 0; r < 8; ++r) wsums[wave][r + 8 * hi] = lsum[r];
    }
    // ---- rescale running output
#pragma unroll
    for (int f = 0; f < 8; ++f)
#pragma unroll
      for (int r = 0; r < 8; ++r) O[f][r] *= alpha[r];
    __syncthreads();
#pragma unroll
    for (int r = 0; r < 8; ++r) {
      int row = r + 8 * hi;
      gsum[r] = gsum[r] * alpha[r] +
                wsums[0][row] + wsums[1][row] + wsums[2][row] + wsums[3][row];
    }
    // ---- O += P * x  over this chunk; wave's d-slice = [wave*128, +128)
#pragma unroll 2
    for (int f = 0; f < 8; ++f) {
      const unsigned short* xb =
          XT + ((size_t)b * D_ + wave * 128 + f * 16) * N_ + (size_t)c * 128;
#pragma unroll
      for (int kc = 0; kc < 4; ++kc) {
        v16bf a = frag_a_rowmajor(sP, LDP_, kc * 32, lane);
        v16bf bx = frag_b_nt(xb, N_, kc * 32, lane);
        O[f] = wmma_bf16(a, bx, O[f]);
      }
    }
    __syncthreads();  // protect sP / wmaxs / wsums before next chunk
  }

  // ---- epilogue: normalize, store bf16 (and fp32 on last iteration)
#pragma unroll
  for (int r = 0; r < 8; ++r) {
    int grow = mt * 16 + r + 8 * hi;
    float inv = 1.0f / gsum[r];
#pragma unroll
    for (int f = 0; f < 8; ++f) {
      float v = O[f][r] * inv;
      size_t idx = ((size_t)b * N_ + grow) * D_ + wave * 128 + f * 16 + n;
      Yb[idx] = f2bf(v);
      if (outF) outF[idx] = v;
    }
  }
}

// ---------------------------------------------------------------------------
extern "C" void kernel_launch(void* const* d_in, const int* in_sizes, int n_in,
                              void* d_out, int out_size, void* d_ws, size_t ws_size,
                              hipStream_t stream) {
  const float* x  = (const float*)d_in[0];
  const float* wq = (const float*)d_in[1];
  const float* wk = (const float*)d_in[2];
  float* out = (float*)d_out;

  char* ws = (char*)d_ws;
  const size_t bnd = (size_t)B_ * N_ * D_ * sizeof(unsigned short);  // 8 MiB
  unsigned short* Xb  = (unsigned short*)(ws + 0 * bnd);
  unsigned short* XT  = (unsigned short*)(ws + 1 * bnd);
  unsigned short* Kb  = (unsigned short*)(ws + 2 * bnd);
  unsigned short* Qb  = (unsigned short*)(ws + 3 * bnd);
  unsigned short* Yb  = (unsigned short*)(ws + 4 * bnd);
  unsigned short* WQb = (unsigned short*)(ws + 5 * bnd);
  unsigned short* WKb = (unsigned short*)(ws + 5 * bnd + (size_t)D_ * D_ * 2);

  prep_kernel<<<4096, 256, 0, stream>>>(x, wq, wk, Xb, XT, WQb, WKb);

  // K = x @ W_K^T  (once)
  proj_kernel<<<(B_ * N_ / 16) * (D_ / 16) / 8, 256, 0, stream>>>(Xb, WKb, Kb);

  const unsigned short* cur = Xb;  // y starts as x
  for (int it = 0; it < ITERS_; ++it) {
    proj_kernel<<<(B_ * N_ / 16) * (D_ / 16) / 8, 256, 0, stream>>>(cur, WQb, Qb);
    flash_kernel<<<B_ * (N_ / 16), 128, 0, stream>>>(
        Kb, Qb, XT, Yb, (it == ITERS_ - 1) ? out : nullptr);
    cur = Yb;
  }
}
